// MultiConvAttn_83279415870164
// MI455X (gfx1250) — matrix operand order repaired and checked by reading, not compiled, not run on hardware
//
#include <hip/hip_runtime.h>
#include <hip/hip_bf16.h>
#include <stdint.h>

// Problem constants (match reference)
#define BB 16
#define SS 2048
#define DD 1024
#define HH 16
#define KK 31
#define NN 496           // H*K
#define MT 64            // positions per workgroup
#define KBLK 32          // D/32 k-steps
#define QSTRIDE 1032     // bf16 elems per LDS row (1024 + 8 pad -> no bank conflicts)
#define SMEM_BYTES (MT * QSTRIDE * 2 + MT * HH * 4)   // 132096 + 4096 = 136192

typedef __attribute__((ext_vector_type(16))) __bf16 bf16x16;
typedef __attribute__((ext_vector_type(8)))  float  f32x8;

union FragU { uint4 u[2]; bf16x16 v; };

__device__ __forceinline__ unsigned short f2bf(float f) {
  unsigned u = __float_as_uint(f);
  u += 0x7FFFu + ((u >> 16) & 1u);       // round-to-nearest-even
  return (unsigned short)(u >> 16);
}

// ---------------------------------------------------------------------------
// Kernel 1: ksum[b,s,h] = sum over 64 channels of key[b,s,h*64 : h*64+64]
// ---------------------------------------------------------------------------
__global__ __launch_bounds__(256) void ksum_kernel(const float* __restrict__ key,
                                                   float* __restrict__ ksum) {
  __shared__ float part[256];
  const int s = blockIdx.x, b = blockIdx.y, tid = threadIdx.x;
  const size_t base = ((size_t)b * SS + s) * DD;
  const float4 v = ((const float4*)(key + base))[tid];   // channels 4t..4t+3
  part[tid] = v.x + v.y + v.z + v.w;
  __syncthreads();
  if (tid < HH) {
    float acc = 0.f;
#pragma unroll
    for (int q = 0; q < 16; ++q) acc += part[tid * 16 + q];   // 16*4 = 64 channels
    ksum[((size_t)b * SS + s) * HH + tid] = acc;
  }
}

// ---------------------------------------------------------------------------
// Kernel 2: repack W_q (D x 496, f32 row-major) into bf16 B-fragment layout:
// packB[((nt*32 + kb)*32 + lane)*16 + e] = bf16(W_q[kg*496 + ng])
//   kg = kb*32 + (e>=8 ? 16 : 0) + (lane>=16 ? 8 : 0) + (e & 7) ,  ng = nt*16 + (lane & 15)
// ---------------------------------------------------------------------------
__global__ __launch_bounds__(256) void packB_kernel(const float* __restrict__ Wq,
                                                    unsigned short* __restrict__ packB) {
  const int f = blockIdx.x * 256 + threadIdx.x;
  if (f >= 31 * KBLK * 32 * 16) return;
  const int e    = f & 15;
  const int lane = (f >> 4) & 31;
  const int kb   = (f >> 9) & 31;
  const int nt   = f >> 14;
  const int kg = kb * 32 + ((e >> 3) << 4) + ((lane >> 4) << 3) + (e & 7);
  const int ng = nt * 16 + (lane & 15);
  packB[f] = f2bf(Wq[(size_t)kg * NN + ng]);
}

// ---------------------------------------------------------------------------
// Kernel 3: fused  weights = Q@W_q + b_q  (bf16 WMMA, f32 acc)  then
// out[b,h,t] = sum_j weights[b,t,h,j] * ksum[b, t+j-15, h] + conv_bias[h]
// 256 thr = 8 waves; 64 positions x full N=496 per workgroup.
// Jobs pair two M-tiles so each B fragment feeds two WMMAs (2 indep chains).
// ---------------------------------------------------------------------------
__global__ __launch_bounds__(256, 2) void fused_gemm_conv(
    const float* __restrict__ q, const float* __restrict__ ksum,
    const unsigned short* __restrict__ packB, const float* __restrict__ b_q,
    const float* __restrict__ conv_bias, float* __restrict__ out) {
  extern __shared__ __align__(16) char dynsmem[];
  unsigned short* qtile = (unsigned short*)dynsmem;                 // [64][QSTRIDE] bf16
  float* outAcc = (float*)(dynsmem + MT * QSTRIDE * 2);             // [64][16] f32

  const int tid = threadIdx.x;
  const int b = blockIdx.y;
  const int p0 = blockIdx.x * MT;

  // ---- Phase A: stage Q tile (f32 -> bf16) into LDS; zero the conv accum ----
  {
    const float4* q4 = (const float4*)(q + ((size_t)b * SS + p0) * DD);
#pragma unroll 4
    for (int i = tid; i < MT * (DD / 4); i += 256) {
      const int row = i >> 8;          // 256 float4 per row
      const int c4  = i & 255;
      const float4 v = q4[(size_t)row * 256 + c4];
      uint2 pk;
      pk.x = (unsigned)f2bf(v.x) | ((unsigned)f2bf(v.y) << 16);
      pk.y = (unsigned)f2bf(v.z) | ((unsigned)f2bf(v.w) << 16);
      *(uint2*)(qtile + row * QSTRIDE + c4 * 4) = pk;
    }
    for (int i = tid; i < MT * HH; i += 256) outAcc[i] = 0.f;
  }
  __syncthreads();

  // ---- Phase B: 2 mt-pairs x 31 n-tiles = 62 jobs round-robined over 8 waves ----
  const int wave  = tid >> 5;
  const int lane  = tid & 31;
  const int mlane = lane & 15;
  const int hsel  = lane >> 4;              // 0: low half, 1: high half
  const int koff  = hsel << 3;              // A-frag K offset (0 or 8)

  for (int job = wave; job < 2 * 31; job += 8) {
    const int mp = job & 1;                 // m-tile pair: rows [mp*32, mp*32+32)
    const int nt = job >> 1;
    const unsigned short* arow0 = qtile + (mp * 32 + mlane) * QSTRIDE + koff;
    const unsigned short* arow1 = arow0 + 16 * QSTRIDE;
    const unsigned short* bbase = packB + ((size_t)(nt * KBLK) * 32 + lane) * 16;
    // warm L2->nearer caches for the *next* job's B block
    __builtin_prefetch((const void*)(bbase + (size_t)4 * KBLK * 512), 0, 2);

    f32x8 acc0 = {}, acc1 = {};
#pragma unroll 4
    for (int kb = 0; kb < KBLK; ++kb) {
      FragU a0, a1, bf;
      a0.u[0] = *(const uint4*)(arow0 + kb * 32);
      a0.u[1] = *(const uint4*)(arow0 + kb * 32 + 16);
      a1.u[0] = *(const uint4*)(arow1 + kb * 32);
      a1.u[1] = *(const uint4*)(arow1 + kb * 32 + 16);
      const unsigned short* bp = bbase + (size_t)kb * 512;
      bf.u[0] = *(const uint4*)bp;
      bf.u[1] = *(const uint4*)(bp + 8);
      acc0 = __builtin_amdgcn_wmma_f32_16x16x32_bf16(false, a0.v, false, bf.v,
                                                     (short)0, acc0, false, false);
      acc1 = __builtin_amdgcn_wmma_f32_16x16x32_bf16(false, a1.v, false, bf.v,
                                                     (short)0, acc1, false, false);
    }

    // ---- epilogue: fold b_q, multiply by shifted ksum taps, LDS-accumulate ----
    // Branchless: clamp the tap index (always-valid load) + select-to-zero.
    const int col = nt * 16 + mlane;          // 0..495
    const int h = col / KK;
    const int j = col - h * KK;               // 0..30
    const float bias = b_q[col];
    const float* ksb = ksum + ((size_t)b * SS) * HH + h;   // index with t*HH
#pragma unroll
    for (int r = 0; r < 8; ++r) {
      const int pl0 = mp * 32 + (hsel << 3) + r;           // C layout: M=r (+8 hi half)
      const int pl1 = pl0 + 16;
      const int t0 = p0 + pl0 + j - 15;                    // pad_l = 15
      const int t1 = t0 + 16;
      const int tc0 = t0 < 0 ? 0 : (t0 > SS - 1 ? SS - 1 : t0);
      const int tc1 = t1 < 0 ? 0 : (t1 > SS - 1 ? SS - 1 : t1);
      const float kraw0 = ksb[tc0 * HH];                   // unconditional, clamped
      const float kraw1 = ksb[tc1 * HH];
      const float kv0 = ((unsigned)t0 < (unsigned)SS) ? kraw0 : 0.f;
      const float kv1 = ((unsigned)t1 < (unsigned)SS) ? kraw1 : 0.f;
      atomicAdd(&outAcc[(pl0 << 4) + h], (acc0[r] + bias) * kv0);
      atomicAdd(&outAcc[(pl1 << 4) + h], (acc1[r] + bias) * kv1);
    }
  }
  __syncthreads();

  // ---- Phase C: out[b,h,p] = acc + conv_bias[h]; single writer per element ----
  for (int i = tid; i < MT * HH; i += 256) {
    const int pl = i >> 4;
    const int h  = i & 15;
    out[((size_t)b * HH + h) * SS + p0 + pl] = outAcc[i] + conv_bias[h];
  }
}

// ---------------------------------------------------------------------------
extern "C" void kernel_launch(void* const* d_in, const int* in_sizes, int n_in,
                              void* d_out, int out_size, void* d_ws, size_t ws_size,
                              hipStream_t stream) {
  const float* q         = (const float*)d_in[0];
  const float* key       = (const float*)d_in[1];
  // d_in[2] (value_sequences) and d_in[3] (lengths) are unused by the reference.
  const float* Wq        = (const float*)d_in[4];
  const float* b_q       = (const float*)d_in[5];
  const float* conv_bias = (const float*)d_in[6];
  float* out = (float*)d_out;

  float* ksum = (float*)d_ws;                                        // 2 MB
  unsigned short* packB = (unsigned short*)((char*)d_ws + (size_t)BB * SS * HH * 4);

  ksum_kernel<<<dim3(SS, BB), 256, 0, stream>>>(key, ksum);
  packB_kernel<<<(31 * KBLK * 32 * 16 + 255) / 256, 256, 0, stream>>>(Wq, packB);
  fused_gemm_conv<<<dim3(SS / MT, BB), 256, SMEM_BYTES, stream>>>(
      q, ksum, packB, b_q, conv_bias, out);
}
